// SelfAwareAttention_3238405341368
// MI455X (gfx1250) — matrix operand order repaired and checked by reading, not compiled, not run on hardware
//
#include <hip/hip_runtime.h>
#include <hip/hip_bf16.h>
#include <cmath>
#include <cstdint>

// ---------------------------------------------------------------------------
// SelfAwareAttention for MI455X (gfx1250), wave32 + WMMA f32_16x16x32_f16,
// with gfx1250 async global->LDS staging (ASYNCcnt) and double buffering.
//   1) proj_kernel    : P[b, 0:1088, HW] = concat(q*scale, k, v, m, n, wf) f16
//   2) tsa_kernel     : flash attention per (b, head), dk=32  -> tsa (f32 ws)
//   3) gsa_kernel     : flash attention, Cr=32 scores, dv=256 -> gsa (f32 ws)
//   4) outproj_kernel : out = g_tsa*tsa + g_gsa*(wo@gsa + bo) + x
// ---------------------------------------------------------------------------

typedef __attribute__((ext_vector_type(16))) _Float16 v16h;
typedef __attribute__((ext_vector_type(8)))  _Float16 v8h;
typedef __attribute__((ext_vector_type(8)))  float    v8f;
typedef __attribute__((ext_vector_type(8)))  float    f8v;

#define HW_       2304
#define C_        256
#define NHEAD     8
#define DK        32
#define PROJ_ROWS 1088
#define NBATCH    2

__device__ __forceinline__ v8f wmma_f16(v16h a, v16h b, v8f c) {
  return __builtin_amdgcn_wmma_f32_16x16x32_f16(false, a, false, b, (short)0, c,
                                                false, false);
}

// gfx1250 async DMA: copy 16B global -> LDS, tracked by ASYNCcnt.
__device__ __forceinline__ void async_ld_b128(unsigned lds_off, const void* gaddr) {
  asm volatile("global_load_async_to_lds_b128 %0, %1, off"
               :: "v"(lds_off), "v"(gaddr) : "memory");
}
__device__ __forceinline__ void wait_async0() {
  asm volatile("s_wait_asynccnt 0x0" ::: "memory");
}
__device__ __forceinline__ unsigned lds_off_of(const void* p) {
  // LDS aperture: addr[31:0] is the workgroup-relative LDS byte offset.
  return (unsigned)(uintptr_t)p;
}

// -------------------------------------------------------------------- proj --
// Block = 128 threads (4 waves). All 4 waves share one 16-column tile of x,
// staged via async-to-LDS; each wave owns one 16-row weight tile.
__global__ void __launch_bounds__(128)
proj_kernel(const float* __restrict__ x,
            const float* __restrict__ wq, const float* __restrict__ bq,
            const float* __restrict__ wk, const float* __restrict__ bk,
            const float* __restrict__ wv, const float* __restrict__ bv,
            const float* __restrict__ wm, const float* __restrict__ bm,
            const float* __restrict__ wn, const float* __restrict__ bn,
            const float* __restrict__ ww, const float* __restrict__ bw,
            _Float16* __restrict__ P)
{
  __shared__ float sX[2][32 * 16];                 // [c 0..31][n 0..15]

  const int t    = threadIdx.x;
  const int lane = t & 31;
  const int wave = t >> 5;                         // 0..3
  const int ctiles  = HW_ / 16;                    // 144
  const int rgroups = (PROJ_ROWS / 16) / 4;        // 17
  const int b  = blockIdx.x / (ctiles * rgroups);
  int rem      = blockIdx.x % (ctiles * rgroups);
  const int ct = rem / rgroups;
  const int rg = rem % rgroups;
  const int r0 = (rg * 4 + wave) * 16;             // 0..1072
  const int n0 = ct * 16;

  const float* wseg; const float* bseg; int rloc; float oscale = 1.0f;
  if      (r0 < 256) { wseg = wq; bseg = bq; rloc = r0;       oscale = 0.1767766952966369f; }
  else if (r0 < 512) { wseg = wk; bseg = bk; rloc = r0 - 256; }
  else if (r0 < 768) { wseg = wv; bseg = bv; rloc = r0 - 512; }
  else if (r0 < 800) { wseg = wm; bseg = bm; rloc = r0 - 768; }
  else if (r0 < 832) { wseg = wn; bseg = bn; rloc = r0 - 800; }
  else               { wseg = ww; bseg = bw; rloc = r0 - 832; }

  const int half = lane >> 4;
  const int l16  = lane & 15;
  const float* xb = x + (size_t)b * C_ * HW_;

  // staging role: thread t copies 16B (4 floats) of row (t>>2), seg (t&3)
  const int srow = t >> 2;                         // 0..31
  const int scol = (t & 3) * 4;                    // 0,4,8,12
  auto stage = [&](int kk, int buf) {
    async_ld_b128(lds_off_of(&sX[buf][srow * 16 + scol]),
                  xb + (size_t)(kk + srow) * HW_ + n0 + scol);
  };

  stage(0, 0);
  v8f acc = {};
  for (int it = 0; it < C_ / 32; ++it) {
    const int kk  = it * 32;
    const int cur = it & 1;
    wait_async0();
    __syncthreads();
    if (it + 1 < C_ / 32) stage(kk + 32, cur ^ 1);

    // A = weight tile (global, 32B contiguous per lane)
    const float* wrow = wseg + (size_t)(rloc + l16) * C_ + kk + half * 8;
    f8v w0 = *(const f8v*)(wrow);
    f8v w1 = *(const f8v*)(wrow + 16);
    v16h a;
    #pragma unroll
    for (int e = 0; e < 8; ++e) { a[e] = (_Float16)w0[e]; a[8 + e] = (_Float16)w1[e]; }
    // B = x tile from LDS: K element e -> c-row (16*(e/8) + half*8 + e%8)
    v16h bf;
    #pragma unroll
    for (int e = 0; e < 16; ++e)
      bf[e] = (_Float16)sX[cur][(((e >> 3) << 4) + half * 8 + (e & 7)) * 16 + l16];
    acc = wmma_f16(a, bf, acc);
  }

  _Float16* Pb = P + ((size_t)b * PROJ_ROWS + r0) * HW_;
  #pragma unroll
  for (int i = 0; i < 8; ++i) {
    int m = i + half * 8;
    float v = (acc[i] + bseg[rloc + m]) * oscale;
    Pb[(size_t)m * HW_ + n0 + l16] = (_Float16)v;
  }
}

// --------------------------------------------------------------------- TSA --
// Block = 256 threads (8 waves), all on the same (b,h); K/V 32x32 f16 tiles
// are staged once per block via async-to-LDS, double buffered.
__global__ void __launch_bounds__(256)
tsa_kernel(const _Float16* __restrict__ P, float* __restrict__ tsa)
{
  __shared__ _Float16 sK[2][32 * 32];
  __shared__ _Float16 sV[2][32 * 32];

  const int t    = threadIdx.x;
  const int lane = t & 31;
  const int wid  = (blockIdx.x * blockDim.x + t) >> 5;
  const int nt_per = HW_ / 16;                     // 144
  const int b  = wid / (NHEAD * nt_per);
  int rem      = wid % (NHEAD * nt_per);
  const int h  = rem / nt_per;
  const int nt = rem % nt_per;
  const int n0 = nt * 16;
  const int half = lane >> 4;
  const int l16  = lane & 15;
  const bool lo  = (half == 0);

  const _Float16* qb = P + ((size_t)b * PROJ_ROWS +   0 + h * DK) * HW_;
  const _Float16* kb = P + ((size_t)b * PROJ_ROWS + 256 + h * DK) * HW_;  // block-uniform
  const _Float16* vb = P + ((size_t)b * PROJ_ROWS + 512 + h * DK) * HW_;  // block-uniform

  // staging role: t<128 -> K tile, else V tile; 16B per thread per chunk
  const int swhich = t >> 7;
  const int sidx   = t & 127;
  const int srow   = sidx >> 2;                    // d 0..31
  const int scol   = (sidx & 3) * 8;               // halfs 0,8,16,24
  const _Float16* sgb = (swhich ? vb : kb) + (size_t)srow * HW_ + scol;
  auto stage = [&](int mc, int buf) {
    unsigned lp = lds_off_of(swhich ? &sV[buf][srow * 32 + scol]
                                    : &sK[buf][srow * 32 + scol]);
    async_ld_b128(lp, sgb + mc);
  };

  // Q fragment (B operand of score WMMA): N = n, K element e -> d = 16*(e/8)+8h+e%8
  v16h qf;
  #pragma unroll
  for (int e = 0; e < 16; ++e)
    qf[e] = qb[(size_t)(((e >> 3) << 4) + half * 8 + (e & 7)) * HW_ + n0 + l16];

  float mrow = -INFINITY, lsum = 0.0f;
  v8f acc0 = {}, acc1 = {};

  stage(0, 0);
  for (int it = 0; it < HW_ / 32; ++it) {
    const int cur = it & 1;
    wait_async0();
    __syncthreads();
    if (it + 1 < HW_ / 32) stage((it + 1) * 32, cur ^ 1);

    // A = K^T tiles from LDS: M = m_local, K element e -> d = 16*(e/8)+8h+e%8
    v16h ak0, ak1;
    #pragma unroll
    for (int e = 0; e < 16; ++e) {
      int d = ((e >> 3) << 4) + half * 8 + (e & 7);
      ak0[e] = sK[cur][d * 32 + l16];
      ak1[e] = sK[cur][d * 32 + 16 + l16];
    }
    v8f z = {};
    v8f s0 = wmma_f16(ak0, qf, z);   // S^T[m 0..15 , n]
    v8f s1 = wmma_f16(ak1, qf, z);   // S^T[m 16..31, n]

    float tmax = s0[0];
    #pragma unroll
    for (int i = 0; i < 8; ++i) { tmax = fmaxf(tmax, s0[i]); tmax = fmaxf(tmax, s1[i]); }
    tmax = fmaxf(tmax, __shfl_xor(tmax, 16));
    float nmax = fmaxf(mrow, tmax);
    float fcorr = __expf(mrow - nmax);
    mrow = nmax;

    float ssum = 0.0f;
    #pragma unroll
    for (int i = 0; i < 8; ++i) {
      s0[i] = __expf(s0[i] - nmax);
      s1[i] = __expf(s1[i] - nmax);
      ssum += s0[i] + s1[i];
    }
    ssum += __shfl_xor(ssum, 16);
    lsum = lsum * fcorr + ssum;
    #pragma unroll
    for (int i = 0; i < 8; ++i) { acc0[i] *= fcorr; acc1[i] *= fcorr; }

    // Repack P^T (C layout) into B-operand layout (K = m_local, N = n)
    v16h pv;
    #pragma unroll
    for (int e = 0; e < 8; ++e) {
      float xr = __shfl_xor(s1[e], 16);
      pv[e] = (_Float16)(lo ? s0[e] : xr);
    }
    #pragma unroll
    for (int e = 0; e < 8; ++e) {
      float xr = __shfl_xor(s0[e], 16);
      pv[8 + e] = (_Float16)(lo ? xr : s1[e]);
    }

    // A = V tiles from LDS (16B ds loads)
    const v8h* vp0 = (const v8h*)&sV[cur][l16 * 32 + half * 16];
    const v8h* vp1 = (const v8h*)&sV[cur][(16 + l16) * 32 + half * 16];
    v8h a0l = vp0[0], a0h = vp0[1];
    v8h a1l = vp1[0], a1h = vp1[1];
    v16h av0, av1;
    #pragma unroll
    for (int e = 0; e < 8; ++e) {
      av0[e] = a0l[e]; av0[8 + e] = a0h[e];
      av1[e] = a1l[e]; av1[8 + e] = a1h[e];
    }
    acc0 = wmma_f16(av0, pv, acc0);
    acc1 = wmma_f16(av1, pv, acc1);
  }

  float rl = 1.0f / lsum;
  float* tb = tsa + ((size_t)b * C_ + h * DK) * HW_ + n0 + l16;
  #pragma unroll
  for (int i = 0; i < 8; ++i) {
    int d = i + half * 8;
    tb[(size_t)d * HW_]        = acc0[i] * rl;
    tb[(size_t)(16 + d) * HW_] = acc1[i] * rl;
  }
}

// --------------------------------------------------------------------- GSA --
// Block = 256 threads (8 waves) = 2 query tiles x 4 channel chunks, same b.
// Keys tile (32x32) and full wf values slab (256x32) staged via async-to-LDS.
__global__ void __launch_bounds__(256)
gsa_kernel(const _Float16* __restrict__ P, float* __restrict__ gsa)
{
  __shared__ _Float16 sK[2][32 * 32];              //  4 KB
  __shared__ _Float16 sV[2][256 * 32];             // 32 KB

  const int t    = threadIdx.x;
  const int lane = t & 31;
  const int wid  = (blockIdx.x * blockDim.x + t) >> 5;
  const int nt_per = HW_ / 16;                     // 144
  const int CHUNKS = 4;
  const int b  = wid / (nt_per * CHUNKS);
  int rem      = wid % (nt_per * CHUNKS);
  const int nt = rem / CHUNKS;
  const int dc = rem % CHUNKS;
  const int n0 = nt * 16;
  const int d0 = dc * 64;
  const int half = lane >> 4;
  const int l16  = lane & 15;
  const bool lo  = (half == 0);

  const _Float16* mb = P + ((size_t)b * PROJ_ROWS + 768) * HW_;   // queries
  const _Float16* nb = P + ((size_t)b * PROJ_ROWS + 800) * HW_;   // keys (uniform)
  const _Float16* wb = P + ((size_t)b * PROJ_ROWS + 832) * HW_;   // values (uniform)

  auto stage = [&](int mc, int buf) {
    if (t < 128) {   // keys: 2KB
      async_ld_b128(lds_off_of(&sK[buf][(t >> 2) * 32 + (t & 3) * 8]),
                    nb + (size_t)(t >> 2) * HW_ + mc + (t & 3) * 8);
    }
    #pragma unroll
    for (int j = 0; j < 4; ++j) {   // values: thread t stages row t (64B)
      async_ld_b128(lds_off_of(&sV[buf][t * 32 + j * 8]),
                    wb + (size_t)t * HW_ + mc + j * 8);
    }
  };

  v16h qf;
  #pragma unroll
  for (int e = 0; e < 16; ++e)
    qf[e] = mb[(size_t)(((e >> 3) << 4) + half * 8 + (e & 7)) * HW_ + n0 + l16];

  float mrow = -INFINITY, lsum = 0.0f;
  v8f acc0 = {}, acc1 = {}, acc2 = {}, acc3 = {};

  stage(0, 0);
  for (int it = 0; it < HW_ / 32; ++it) {
    const int cur = it & 1;
    wait_async0();
    __syncthreads();
    if (it + 1 < HW_ / 32) stage((it + 1) * 32, cur ^ 1);

    v16h ak0, ak1;
    #pragma unroll
    for (int e = 0; e < 16; ++e) {
      int d = ((e >> 3) << 4) + half * 8 + (e & 7);
      ak0[e] = sK[cur][d * 32 + l16];
      ak1[e] = sK[cur][d * 32 + 16 + l16];
    }
    v8f z = {};
    v8f s0 = wmma_f16(ak0, qf, z);
    v8f s1 = wmma_f16(ak1, qf, z);

    float tmax = s0[0];
    #pragma unroll
    for (int i = 0; i < 8; ++i) { tmax = fmaxf(tmax, s0[i]); tmax = fmaxf(tmax, s1[i]); }
    tmax = fmaxf(tmax, __shfl_xor(tmax, 16));
    float nmax = fmaxf(mrow, tmax);
    float fcorr = __expf(mrow - nmax);
    mrow = nmax;

    float ssum = 0.0f;
    #pragma unroll
    for (int i = 0; i < 8; ++i) {
      s0[i] = __expf(s0[i] - nmax);
      s1[i] = __expf(s1[i] - nmax);
      ssum += s0[i] + s1[i];
    }
    ssum += __shfl_xor(ssum, 16);
    lsum = lsum * fcorr + ssum;
    #pragma unroll
    for (int i = 0; i < 8; ++i) {
      acc0[i] *= fcorr; acc1[i] *= fcorr; acc2[i] *= fcorr; acc3[i] *= fcorr;
    }

    v16h pv;
    #pragma unroll
    for (int e = 0; e < 8; ++e) {
      float xr = __shfl_xor(s1[e], 16);
      pv[e] = (_Float16)(lo ? s0[e] : xr);
    }
    #pragma unroll
    for (int e = 0; e < 8; ++e) {
      float xr = __shfl_xor(s0[e], 16);
      pv[8 + e] = (_Float16)(lo ? xr : s1[e]);
    }

    v8f* accs[4] = { &acc0, &acc1, &acc2, &acc3 };
    #pragma unroll
    for (int tt = 0; tt < 4; ++tt) {
      const v8h* vp = (const v8h*)&sV[cur][(d0 + tt * 16 + l16) * 32 + half * 16];
      v8h alo2 = vp[0], ahi2 = vp[1];
      v16h av;
      #pragma unroll
      for (int e = 0; e < 8; ++e) { av[e] = alo2[e]; av[8 + e] = ahi2[e]; }
      *accs[tt] = wmma_f16(av, pv, *accs[tt]);
    }
  }

  float rl = 1.0f / lsum;
  v8f* accs[4] = { &acc0, &acc1, &acc2, &acc3 };
  #pragma unroll
  for (int tt = 0; tt < 4; ++tt) {
    float* gb = gsa + ((size_t)b * C_ + d0 + tt * 16) * HW_ + n0 + l16;
    #pragma unroll
    for (int i = 0; i < 8; ++i) {
      int d = i + half * 8;
      gb[(size_t)d * HW_] = (*accs[tt])[i] * rl;
    }
  }
}

// ---------------------------------------------------------- output project --
// Block = 128 threads (4 waves) sharing one staged gsa column tile.
__global__ void __launch_bounds__(128)
outproj_kernel(const float* __restrict__ x,  const float* __restrict__ wo,
               const float* __restrict__ bo, const float* __restrict__ tsa,
               const float* __restrict__ gsa,
               const float* __restrict__ g_tsa, const float* __restrict__ g_gsa,
               float* __restrict__ out)
{
  __shared__ float sX[2][32 * 16];

  const int t    = threadIdx.x;
  const int lane = t & 31;
  const int wave = t >> 5;
  const int ctiles  = HW_ / 16;                    // 144
  const int rgroups = (C_ / 16) / 4;               // 4
  const int b  = blockIdx.x / (ctiles * rgroups);
  int rem      = blockIdx.x % (ctiles * rgroups);
  const int ct = rem / rgroups;
  const int rg = rem % rgroups;
  const int r0 = (rg * 4 + wave) * 16;
  const int n0 = ct * 16;
  const int half = lane >> 4;
  const int l16  = lane & 15;

  const float* gb = gsa + (size_t)b * C_ * HW_;

  const int srow = t >> 2;
  const int scol = (t & 3) * 4;
  auto stage = [&](int kk, int buf) {
    async_ld_b128(lds_off_of(&sX[buf][srow * 16 + scol]),
                  gb + (size_t)(kk + srow) * HW_ + n0 + scol);
  };

  stage(0, 0);
  v8f acc = {};
  for (int it = 0; it < C_ / 32; ++it) {
    const int kk  = it * 32;
    const int cur = it & 1;
    wait_async0();
    __syncthreads();
    if (it + 1 < C_ / 32) stage(kk + 32, cur ^ 1);

    const float* wrow = wo + (size_t)(r0 + l16) * C_ + kk + half * 8;
    f8v w0 = *(const f8v*)(wrow);
    f8v w1 = *(const f8v*)(wrow + 16);
    v16h a;
    #pragma unroll
    for (int e = 0; e < 8; ++e) { a[e] = (_Float16)w0[e]; a[8 + e] = (_Float16)w1[e]; }
    v16h bf;
    #pragma unroll
    for (int e = 0; e < 16; ++e)
      bf[e] = (_Float16)sX[cur][(((e >> 3) << 4) + half * 8 + (e & 7)) * 16 + l16];
    acc = wmma_f16(a, bf, acc);
  }

  float gt = g_tsa[0], gg = g_gsa[0];
  #pragma unroll
  for (int i = 0; i < 8; ++i) {
    int row = r0 + i + half * 8;
    size_t idx = ((size_t)b * C_ + row) * HW_ + n0 + l16;
    out[idx] = gt * tsa[idx] + gg * (acc[i] + bo[row]) + x[idx];
  }
}

// ------------------------------------------------------------------ launch --
extern "C" void kernel_launch(void* const* d_in, const int* in_sizes, int n_in,
                              void* d_out, int out_size, void* d_ws, size_t ws_size,
                              hipStream_t stream) {
  const float* x  = (const float*)d_in[0];
  const float* wq = (const float*)d_in[1];  const float* bq = (const float*)d_in[2];
  const float* wk = (const float*)d_in[3];  const float* bk = (const float*)d_in[4];
  const float* wv = (const float*)d_in[5];  const float* bv = (const float*)d_in[6];
  const float* wm = (const float*)d_in[7];  const float* bm = (const float*)d_in[8];
  const float* wn = (const float*)d_in[9];  const float* bn = (const float*)d_in[10];
  const float* ww = (const float*)d_in[11]; const float* bw = (const float*)d_in[12];
  const float* wo = (const float*)d_in[13]; const float* bo = (const float*)d_in[14];
  const float* gt = (const float*)d_in[15]; const float* gg = (const float*)d_in[16];
  float* out = (float*)d_out;

  char* ws = (char*)d_ws;
  _Float16* P = (_Float16*)ws;                                     // ~10.0 MB f16
  size_t pbytes = (size_t)NBATCH * PROJ_ROWS * HW_ * sizeof(_Float16);
  float* tsa = (float*)(ws + pbytes);                              // ~4.7 MB f32
  size_t tbytes = (size_t)NBATCH * C_ * HW_ * sizeof(float);
  float* gsa = (float*)(ws + pbytes + tbytes);                     // ~4.7 MB f32

  int blkA = NBATCH * (HW_ / 16) * ((PROJ_ROWS / 16) / 4);   // 4896 blocks x 128
  proj_kernel<<<blkA, 128, 0, stream>>>(x, wq, bq, wk, bk, wv, bv,
                                        wm, bm, wn, bn, ww, bw, P);
  int blkB = NBATCH * NHEAD * (HW_ / 16) / 8;                // 288 blocks x 256
  tsa_kernel<<<blkB, 256, 0, stream>>>(P, tsa);
  int blkC = NBATCH * (HW_ / 16) * 4 / 8;                    // 144 blocks x 256
  gsa_kernel<<<blkC, 256, 0, stream>>>(P, gsa);
  int blkD = NBATCH * (HW_ / 16) * ((C_ / 16) / 4);          // 1152 blocks x 128
  outproj_kernel<<<blkD, 128, 0, stream>>>(x, wo, bo, tsa, gsa, gt, gg, out);
}
